// TransformerEncoderLayer_66778151518243
// MI455X (gfx1250) — compile-verified, matrix operand order checked by reading, not run on hardware
//
#include <hip/hip_runtime.h>
#include <hip/hip_bf16.h>

// ---------------- problem constants (match reference) ----------------
#define DMODEL 1024
#define NHEADS 16
#define DK     64
#define DFF    4096
#define SEQ    2048
#define BATCH  4
#define MTOK   (BATCH * SEQ)          // 8192 token rows
#define NEGBIG (-1.0e9f)

typedef __bf16 bf16_t;
typedef __attribute__((ext_vector_type(16))) __bf16 v16bf;
typedef __attribute__((ext_vector_type(8)))  __bf16 v8bf;
typedef __attribute__((ext_vector_type(8)))  float  v8f;
typedef int v4i_t __attribute__((vector_size(16)));   // v4i32 (async-LDS payload)

// ---- CDNA5 async global->LDS path (probe via __has_builtin) ----------------
#if defined(__has_builtin)
#if __has_builtin(__builtin_amdgcn_global_load_async_to_lds_b128)
#define HAS_ASYNC_LDS 1
#endif
#endif
#ifndef HAS_ASYNC_LDS
#define HAS_ASYNC_LDS 0
#endif

union BF16x16 { v16bf v; v8bf h[2]; };

// Load one 16-lane-striped bf16 fragment: 8 contiguous elems at p, 8 at p+16.
// Caller computes p = row_base + kg*8 per the ISA 16-bit A/B VGPR layout.
__device__ __forceinline__ v16bf frag_load(const bf16_t* p) {
  BF16x16 u;
  u.h[0] = *(const v8bf*)(p);
  u.h[1] = *(const v8bf*)(p + 16);
  return u.v;
}

__device__ __forceinline__ v8f wmma_bf(v16bf a, v16bf b, v8f c) {
  return __builtin_amdgcn_wmma_f32_16x16x32_bf16(
      false, a, false, b, (short)0, c, false, false);
}

__device__ __forceinline__ v8f v8f_zero() {
  v8f z;
#pragma unroll
  for (int i = 0; i < 8; ++i) z[i] = 0.0f;
  return z;
}

// 16-byte global -> LDS copy; async DMA when available, else via VGPRs.
__device__ __forceinline__ void copy16_g2l(const bf16_t* g, bf16_t* l) {
#if HAS_ASYNC_LDS
  __builtin_amdgcn_global_load_async_to_lds_b128(
      (__attribute__((address_space(1))) v4i_t*)g,
      (__attribute__((address_space(3))) v4i_t*)l,
      0, 0);
#else
  *(v8bf*)l = *(const v8bf*)g;
#endif
}

// Wait until at most `n` async loads remain outstanding (no-op on sync path).
__device__ __forceinline__ void wait_async_le(int n) {
#if HAS_ASYNC_LDS
#if __has_builtin(__builtin_amdgcn_s_wait_asynccnt)
  if (n == 0) __builtin_amdgcn_s_wait_asynccnt(0);
  else        __builtin_amdgcn_s_wait_asynccnt(4);
#else
  if (n == 0) asm volatile("s_wait_asynccnt 0x0" ::: "memory");
  else        asm volatile("s_wait_asynccnt 0x4" ::: "memory");
#endif
#else
  (void)n;
#endif
}

// ---------------- LayerNorm (f32 in -> bf16 out), one row per block --------
__global__ __launch_bounds__(256) void ln_bf16_kernel(
    const float* __restrict__ x, const float* __restrict__ g,
    const float* __restrict__ beta, bf16_t* __restrict__ out) {
  __shared__ float s_sum[256];
  __shared__ float s_sq[256];
  const size_t row = blockIdx.x;
  const float* xr = x + row * DMODEL;
  float s = 0.f, sq = 0.f;
#pragma unroll
  for (int j = threadIdx.x; j < DMODEL; j += 256) {
    float v = xr[j]; s += v; sq += v * v;
  }
  s_sum[threadIdx.x] = s; s_sq[threadIdx.x] = sq;
  __syncthreads();
  for (int off = 128; off > 0; off >>= 1) {
    if (threadIdx.x < (unsigned)off) {
      s_sum[threadIdx.x] += s_sum[threadIdx.x + off];
      s_sq[threadIdx.x]  += s_sq[threadIdx.x + off];
    }
    __syncthreads();
  }
  const float mu  = s_sum[0] * (1.0f / DMODEL);
  const float var = s_sq[0] * (1.0f / DMODEL) - mu * mu;
  const float rs  = rsqrtf(var + 1e-6f);
  bf16_t* orow = out + row * DMODEL;
#pragma unroll
  for (int j = threadIdx.x; j < DMODEL; j += 256)
    orow[j] = (bf16_t)((xr[j] - mu) * rs * g[j] + beta[j]);
}

// -------- weight convert: f32 W[K,N] -> bf16 Wt[N,K] (N-major for B-frags) --
__global__ __launch_bounds__(256) void w_to_bf16t_kernel(
    const float* __restrict__ w, bf16_t* __restrict__ wt, int K, int N) {
  size_t idx = (size_t)blockIdx.x * 256 + threadIdx.x;
  size_t total = (size_t)K * N;
  if (idx >= total) return;
  int n = (int)(idx / K);
  int k = (int)(idx % K);
  wt[idx] = (bf16_t)w[(size_t)k * N + n];
}

// -------- V transpose: bf16 [B*S, D] -> bf16 [B,H,DK,S] -------------------
__global__ __launch_bounds__(256) void v_transpose_kernel(
    const bf16_t* __restrict__ vb, bf16_t* __restrict__ vt) {
  size_t idx = (size_t)blockIdx.x * 256 + threadIdx.x;
  const size_t total = (size_t)BATCH * NHEADS * DK * SEQ;
  if (idx >= total) return;
  int s = (int)(idx % SEQ);
  size_t r = idx / SEQ;
  int d = (int)(r % DK);
  size_t bh = r / DK;
  int h = (int)(bh % NHEADS);
  int b = (int)(bh / NHEADS);
  vt[idx] = vb[((size_t)b * SEQ + s) * DMODEL + h * DK + d];
}

// ---------------- bf16 WMMA GEMM with async double-buffered LDS -------------
// C[M,N] = A[M,K](bf16,row-major) * Bt[N,K](bf16,N-major) (+bias)(+relu)(+res)
// Block: 256 thr = 8 waves; block tile 128x128; wave tile 64(M)x32(N);
// K staged in 32-wide chunks through LDS (2 buffers, ASYNCcnt-pipelined).
template <bool BIAS, bool RELU, bool RES, bool OUTF, bool OUTB>
__global__ __launch_bounds__(256) void gemm_bf16_kernel(
    const bf16_t* __restrict__ A, const bf16_t* __restrict__ Bt,
    int M, int N, int K,
    const float* __restrict__ bias, const float* __restrict__ residual,
    float* __restrict__ outF, bf16_t* __restrict__ outB, float bscale) {
  __shared__ bf16_t As[2][128 * 32];   // 8KB per buffer
  __shared__ bf16_t Bs[2][128 * 32];   // 8KB per buffer

  const int tid  = threadIdx.x;
  const int lane = tid & 31;
  const int wave = tid >> 5;
  const int wm = wave >> 2;            // 0..1
  const int wn = wave & 3;             // 0..3
  const int mblk = blockIdx.y * 128;
  const int nblk = blockIdx.x * 128;
  const int r  = lane & 15;            // row (A) / col (B) within 16
  const int kg = lane >> 4;            // K-group select (ISA 16-bit layout)

  // Stage one 128x32 A tile + one 128x32 B tile: 4 x b128 per thread.
  auto stage = [&](int buf, int k0) {
#pragma unroll
    for (int j = 0; j < 2; ++j) {
      const int vec = tid + j * 256;        // 0..511
      const int row = vec >> 2;             // 0..127
      const int c   = (vec & 3) << 3;       // 0,8,16,24
      copy16_g2l(A  + (size_t)(mblk + row) * K + k0 + c, &As[buf][row * 32 + c]);
      copy16_g2l(Bt + (size_t)(nblk + row) * K + k0 + c, &Bs[buf][row * 32 + c]);
    }
  };

  v8f acc[4][2];
#pragma unroll
  for (int mi = 0; mi < 4; ++mi)
#pragma unroll
    for (int nj = 0; nj < 2; ++nj) acc[mi][nj] = v8f_zero();

  const int nK = K >> 5;
  int cur = 0;
  stage(0, 0);
  for (int kt = 0; kt < nK; ++kt) {
    const bool has_next = (kt + 1) < nK;
    if (has_next) stage(cur ^ 1, (kt + 1) << 5);
    wait_async_le(has_next ? 4 : 0);   // previous tile landed; next in flight
    __syncthreads();

    v16bf af[4];
#pragma unroll
    for (int mi = 0; mi < 4; ++mi)
      af[mi] = frag_load(&As[cur][(wm * 64 + mi * 16 + r) * 32 + kg * 8]);
    v16bf bfr[2];
#pragma unroll
    for (int nj = 0; nj < 2; ++nj)
      bfr[nj] = frag_load(&Bs[cur][(wn * 32 + nj * 16 + r) * 32 + kg * 8]);
#pragma unroll
    for (int mi = 0; mi < 4; ++mi)
#pragma unroll
      for (int nj = 0; nj < 2; ++nj)
        acc[mi][nj] = wmma_bf(af[mi], bfr[nj], acc[mi][nj]);

    __syncthreads();                   // buffer free before next-next stage
    cur ^= 1;
  }

#pragma unroll
  for (int mi = 0; mi < 4; ++mi)
#pragma unroll
    for (int nj = 0; nj < 2; ++nj)
#pragma unroll
      for (int i = 0; i < 8; ++i) {
        const int rr = mblk + wm * 64 + mi * 16 + i + 8 * kg; // M = i + 8*half
        const int cc = nblk + wn * 32 + nj * 16 + r;          // N = lane & 15
        float v = acc[mi][nj][i];
        if (BIAS) v += bias[cc];
        if (RELU) v = fmaxf(v, 0.0f);
        if (RES)  v += residual[(size_t)rr * N + cc];
        if (OUTF) outF[(size_t)rr * N + cc] = v;
        if (OUTB) outB[(size_t)rr * N + cc] = (bf16_t)(v * bscale);
      }
}

// ---------------- flash attention: one wave per 16-query tile ---------------
// qb: bf16 [B*S, D] pre-scaled by 1/sqrt(dk); kb: bf16 [B*S, D];
// vt: bf16 [B,H,DK,S]; mask: int [B,S,S]; ctx: bf16 [B*S, D].
__global__ __launch_bounds__(32) void attn_kernel(
    const bf16_t* __restrict__ qb, const bf16_t* __restrict__ kb,
    const bf16_t* __restrict__ vt, const int* __restrict__ mask,
    bf16_t* __restrict__ ctx) {
  __shared__ bf16_t Pl[16][72];     // 16x64 P tile, padded row stride (144B)
  const int lane = threadIdx.x;
  const int r  = lane & 15;
  const int kg = lane >> 4;
  const int q0 = blockIdx.x * 16;
  const int h  = blockIdx.y;
  const int b  = blockIdx.z;

  // Q fragments stay resident (dk=64 -> two K-chunks of 32)
  const bf16_t* qbase = qb + ((size_t)(b * SEQ + q0 + r)) * DMODEL + h * DK;
  const v16bf qf0 = frag_load(qbase + kg * 8);
  const v16bf qf1 = frag_load(qbase + 32 + kg * 8);

  float m_run[8], l_run[8];
  v8f acc[4];
#pragma unroll
  for (int i = 0; i < 8; ++i) { m_run[i] = -INFINITY; l_run[i] = 0.0f; }
#pragma unroll
  for (int nj = 0; nj < 4; ++nj) acc[nj] = v8f_zero();

  const int* mb = mask + (size_t)b * SEQ * SEQ;

  for (int k0 = 0; k0 < SEQ; k0 += 64) {
    // ---- scores: 16(q) x 64(k), 4 WMMA tiles, K=dk=64 over 2 chunks ----
    v8f sc[4];
#pragma unroll
    for (int nj = 0; nj < 4; ++nj) {
      const bf16_t* kbase =
          kb + ((size_t)(b * SEQ + k0 + nj * 16 + r)) * DMODEL + h * DK;
      v16bf b0 = frag_load(kbase + kg * 8);
      v16bf b1 = frag_load(kbase + 32 + kg * 8);
      v8f c = v8f_zero();
      c = wmma_bf(qf0, b0, c);
      c = wmma_bf(qf1, b1, c);
      sc[nj] = c;
    }
    // ---- mask (0 -> -1e9) ----
#pragma unroll
    for (int nj = 0; nj < 4; ++nj)
#pragma unroll
      for (int i = 0; i < 8; ++i) {
        int mv = mb[(size_t)(q0 + i + 8 * kg) * SEQ + k0 + nj * 16 + r];
        if (mv == 0) sc[nj][i] = NEGBIG;
      }
    // ---- online softmax: row max / rescale ----
    float mnew[8], f[8];
#pragma unroll
    for (int i = 0; i < 8; ++i) {
      float v = fmaxf(fmaxf(sc[0][i], sc[1][i]), fmaxf(sc[2][i], sc[3][i]));
      v = fmaxf(v, __shfl_xor(v, 1, 32));
      v = fmaxf(v, __shfl_xor(v, 2, 32));
      v = fmaxf(v, __shfl_xor(v, 4, 32));
      v = fmaxf(v, __shfl_xor(v, 8, 32));   // reduce 16 lanes within half
      mnew[i] = fmaxf(m_run[i], v);
      f[i] = (m_run[i] > -1e30f) ? __expf(m_run[i] - mnew[i]) : 0.0f;
    }
    float ts[8];
#pragma unroll
    for (int i = 0; i < 8; ++i) ts[i] = 0.0f;
#pragma unroll
    for (int nj = 0; nj < 4; ++nj)
#pragma unroll
      for (int i = 0; i < 8; ++i) {
        float p = __expf(sc[nj][i] - mnew[i]);
        sc[nj][i] = p;
        ts[i] += p;
      }
#pragma unroll
    for (int i = 0; i < 8; ++i) {
      float t = ts[i];
      t += __shfl_xor(t, 1, 32);
      t += __shfl_xor(t, 2, 32);
      t += __shfl_xor(t, 4, 32);
      t += __shfl_xor(t, 8, 32);
      l_run[i] = l_run[i] * f[i] + t;
      m_run[i] = mnew[i];
    }
#pragma unroll
    for (int nj = 0; nj < 4; ++nj)
#pragma unroll
      for (int i = 0; i < 8; ++i) acc[nj][i] *= f[i];

    // ---- P (C-layout) -> LDS -> A-fragment layout ----
#pragma unroll
    for (int nj = 0; nj < 4; ++nj)
#pragma unroll
      for (int i = 0; i < 8; ++i)
        Pl[i + 8 * kg][nj * 16 + r] = (bf16_t)sc[nj][i];
    __syncthreads();
    v16bf p0 = frag_load(&Pl[r][kg * 8]);
    v16bf p1 = frag_load(&Pl[r][32 + kg * 8]);
    // ---- ctx += P @ V (V transposed: contiguous in token dim) ----
#pragma unroll
    for (int nj2 = 0; nj2 < 4; ++nj2) {
      const bf16_t* vbase =
          vt + ((size_t)((b * NHEADS + h) * DK + nj2 * 16 + r)) * SEQ + k0;
      v16bf v0 = frag_load(vbase + kg * 8);
      v16bf v1 = frag_load(vbase + 32 + kg * 8);
      acc[nj2] = wmma_bf(p0, v0, acc[nj2]);
      acc[nj2] = wmma_bf(p1, v1, acc[nj2]);
    }
    __syncthreads();
  }

  float inv[8];
#pragma unroll
  for (int i = 0; i < 8; ++i)
    inv[i] = (l_run[i] > 0.0f) ? (1.0f / l_run[i]) : 0.0f;
#pragma unroll
  for (int nj2 = 0; nj2 < 4; ++nj2)
#pragma unroll
    for (int i = 0; i < 8; ++i)
      ctx[((size_t)(b * SEQ + q0 + i + 8 * kg)) * DMODEL + h * DK + nj2 * 16 + r] =
          (bf16_t)(acc[nj2][i] * inv[i]);
}

// ---------------- host side -------------------------------------------------
static char* ws_bump(char*& p, size_t bytes) {
  char* ret = p;
  p += (bytes + 255) & ~(size_t)255;
  return ret;
}

extern "C" void kernel_launch(void* const* d_in, const int* in_sizes, int n_in,
                              void* d_out, int out_size, void* d_ws, size_t ws_size,
                              hipStream_t stream) {
  (void)in_sizes; (void)n_in; (void)out_size; (void)ws_size;
  const float* x     = (const float*)d_in[0];
  const int*   mask  = (const int*)d_in[1];
  const float* w_q   = (const float*)d_in[2];
  const float* w_k   = (const float*)d_in[3];
  const float* w_v   = (const float*)d_in[4];
  const float* w_o   = (const float*)d_in[5];
  const float* b_o   = (const float*)d_in[6];
  const float* w1    = (const float*)d_in[7];
  const float* b1    = (const float*)d_in[8];
  const float* w2    = (const float*)d_in[9];
  const float* b2    = (const float*)d_in[10];
  const float* ln1_g = (const float*)d_in[11];
  const float* ln1_b = (const float*)d_in[12];
  const float* ln2_g = (const float*)d_in[13];
  const float* ln2_b = (const float*)d_in[14];
  float* out = (float*)d_out;

  char* p = (char*)d_ws;
  bf16_t* wqt  = (bf16_t*)ws_bump(p, (size_t)DMODEL * DMODEL * 2);
  bf16_t* wkt  = (bf16_t*)ws_bump(p, (size_t)DMODEL * DMODEL * 2);
  bf16_t* wvt  = (bf16_t*)ws_bump(p, (size_t)DMODEL * DMODEL * 2);
  bf16_t* wot  = (bf16_t*)ws_bump(p, (size_t)DMODEL * DMODEL * 2);
  bf16_t* w1t  = (bf16_t*)ws_bump(p, (size_t)DMODEL * DFF * 2);
  bf16_t* w2t  = (bf16_t*)ws_bump(p, (size_t)DFF * DMODEL * 2);
  bf16_t* nx   = (bf16_t*)ws_bump(p, (size_t)MTOK * DMODEL * 2);
  bf16_t* qbuf = (bf16_t*)ws_bump(p, (size_t)MTOK * DMODEL * 2);
  bf16_t* kbuf = (bf16_t*)ws_bump(p, (size_t)MTOK * DMODEL * 2);
  bf16_t* vbuf = (bf16_t*)ws_bump(p, (size_t)MTOK * DMODEL * 2);
  bf16_t* vtb  = (bf16_t*)ws_bump(p, (size_t)MTOK * DMODEL * 2);
  bf16_t* ctxb = (bf16_t*)ws_bump(p, (size_t)MTOK * DMODEL * 2);
  float*  x1   = (float*)ws_bump(p, (size_t)MTOK * DMODEL * 4);
  bf16_t* nx2  = (bf16_t*)ws_bump(p, (size_t)MTOK * DMODEL * 2);
  bf16_t* hb   = (bf16_t*)ws_bump(p, (size_t)MTOK * DFF * 2);

  // 1. weight conversion to bf16, N-major
  {
    int blkDD = (DMODEL * DMODEL + 255) / 256;
    int blkDF = (DMODEL * DFF + 255) / 256;
    w_to_bf16t_kernel<<<blkDD, 256, 0, stream>>>(w_q, wqt, DMODEL, DMODEL);
    w_to_bf16t_kernel<<<blkDD, 256, 0, stream>>>(w_k, wkt, DMODEL, DMODEL);
    w_to_bf16t_kernel<<<blkDD, 256, 0, stream>>>(w_v, wvt, DMODEL, DMODEL);
    w_to_bf16t_kernel<<<blkDD, 256, 0, stream>>>(w_o, wot, DMODEL, DMODEL);
    w_to_bf16t_kernel<<<blkDF, 256, 0, stream>>>(w1, w1t, DMODEL, DFF);
    w_to_bf16t_kernel<<<blkDF, 256, 0, stream>>>(w2, w2t, DFF, DMODEL);
  }
  // 2. LN1
  ln_bf16_kernel<<<MTOK, 256, 0, stream>>>(x, ln1_g, ln1_b, nx);
  // 3. Q/K/V projections (bf16 out; Q pre-scaled by 1/sqrt(dk))
  {
    dim3 grid(DMODEL / 128, MTOK / 128);
    gemm_bf16_kernel<false, false, false, false, true>
        <<<grid, 256, 0, stream>>>(nx, wqt, MTOK, DMODEL, DMODEL,
                                   nullptr, nullptr, nullptr, qbuf, 0.125f);
    gemm_bf16_kernel<false, false, false, false, true>
        <<<grid, 256, 0, stream>>>(nx, wkt, MTOK, DMODEL, DMODEL,
                                   nullptr, nullptr, nullptr, kbuf, 1.0f);
    gemm_bf16_kernel<false, false, false, false, true>
        <<<grid, 256, 0, stream>>>(nx, wvt, MTOK, DMODEL, DMODEL,
                                   nullptr, nullptr, nullptr, vbuf, 1.0f);
  }
  // 4. V transpose to [B,H,DK,S]
  {
    size_t total = (size_t)MTOK * DMODEL;
    v_transpose_kernel<<<(int)((total + 255) / 256), 256, 0, stream>>>(vbuf, vtb);
  }
  // 5. flash attention
  {
    dim3 grid(SEQ / 16, NHEADS, BATCH);
    attn_kernel<<<grid, 32, 0, stream>>>(qbuf, kbuf, vtb, mask, ctxb);
  }
  // 6. output projection + bias + residual(x) -> x1 (f32)
  {
    dim3 grid(DMODEL / 128, MTOK / 128);
    gemm_bf16_kernel<true, false, true, true, false>
        <<<grid, 256, 0, stream>>>(ctxb, wot, MTOK, DMODEL, DMODEL,
                                   b_o, x, x1, nullptr, 1.0f);
  }
  // 7. LN2
  ln_bf16_kernel<<<MTOK, 256, 0, stream>>>(x1, ln2_g, ln2_b, nx2);
  // 8. FFN GEMM1 + bias + relu -> bf16 hidden
  {
    dim3 grid(DFF / 128, MTOK / 128);
    gemm_bf16_kernel<true, true, false, false, true>
        <<<grid, 256, 0, stream>>>(nx2, w1t, MTOK, DFF, DMODEL,
                                   b1, nullptr, nullptr, hb, 1.0f);
  }
  // 9. FFN GEMM2 + bias + residual(x1) -> d_out (f32)
  {
    dim3 grid(DMODEL / 128, MTOK / 128);
    gemm_bf16_kernel<true, false, true, true, false>
        <<<grid, 256, 0, stream>>>(hb, w2t, MTOK, DMODEL, DFF,
                                   b2, x1, out, nullptr, 1.0f);
  }
}
